// LinearSelfAttention_5755256176957
// MI455X (gfx1250) — matrix operand order repaired and checked by reading, not compiled, not run on hardware
//
#include <hip/hip_runtime.h>
#include <hip/hip_bf16.h>

// ---------------------------------------------------------------------------
// LinearSelfAttention for MI455X (gfx1250, wave32, WMMA)
//   x:(8,384,4,4096) f32, w_qkv:(769,384), b_qkv:(769), w_out:(384,384), b_out:(384)
// bf16 WMMA (16x16x32, fp32 accum) for both GEMMs, double-buffered LDS tiles,
// async global->LDS (ASYNCcnt) for weight tiles.
// ---------------------------------------------------------------------------

#define EMBED 384
#define NB    8
#define NP    4
#define NN    4096
#define NSLAB (NB * NP)         // 32
#define OQKV  (1 + 2 * EMBED)   // 769
#define OPAD  784               // 769 padded to 7*112 (49 tiles of 16)
#define KSTEPS (EMBED / 32)     // 12

typedef __bf16 bf16_t;
typedef __attribute__((ext_vector_type(16))) __bf16 v16bf;
typedef __attribute__((ext_vector_type(8)))  __bf16 v8bf;
typedef __attribute__((ext_vector_type(2)))  __bf16 v2bf;
typedef __attribute__((ext_vector_type(8)))  float  v8f;
typedef __attribute__((ext_vector_type(4)))  float  v4f;
typedef __attribute__((ext_vector_type(4)))  unsigned int v4u;
typedef __attribute__((ext_vector_type(4)))  int    v4i;

// address-space-qualified pointer types for the async-copy builtin
typedef __attribute__((address_space(1))) v4i* gas_v4i_p;   // global (AS1)
typedef __attribute__((address_space(3))) v4i* las_v4i_p;   // LDS    (AS3)

// ---------------- async global->LDS copy (ASYNCcnt path) -------------------
__device__ __forceinline__ void async_copy16(const void* g, void* l) {
#if __has_builtin(__builtin_amdgcn_global_load_async_to_lds_b128)
  __builtin_amdgcn_global_load_async_to_lds_b128(
      (gas_v4i_p)(void*)g, (las_v4i_p)l, 0, 0);
#else
  *(v4u*)l = *(const v4u*)g;   // fallback: regular load/store pair
#endif
}

__device__ __forceinline__ void async_wait_all() {
#if __has_builtin(__builtin_amdgcn_global_load_async_to_lds_b128)
#if __has_builtin(__builtin_amdgcn_s_wait_asynccnt)
  __builtin_amdgcn_s_wait_asynccnt(0);
#else
  asm volatile("s_wait_asynccnt 0" ::: "memory");
#endif
#endif
}

// ---------------- WMMA fragment builders (ISA 7.12.2 layouts) --------------
// A (16x32 bf16): lane L (0..15): row M=L, K = {0..7, 16..23}
//                 lane L+16     : row M=L, K = {8..15, 24..31}
__device__ __forceinline__ v16bf load_a_frag(const bf16_t* tile /*row-major, stride 32*/,
                                             int lane) {
  const int r  = lane & 15;
  const int kb = (lane >> 4) << 3;          // 0 or 8
  const bf16_t* p = tile + r * 32 + kb;
  v8bf lo = *(const v8bf*)(p);              // K = kb .. kb+7
  v8bf hi = *(const v8bf*)(p + 16);         // K = kb+16 .. kb+23
  return __builtin_shufflevector(lo, hi, 0,1,2,3,4,5,6,7,8,9,10,11,12,13,14,15);
}

// B (32x16 bf16): lane L (0..15): col N=L,  K = {0..7, 16..23}
//                 lane L+16     : col N=L,  K = {8..15, 24..31}
// tileT is the 128x32 transposed [n][k] LDS image of the K x N operand.
__device__ __forceinline__ v16bf load_b_frag(const bf16_t* tileT, int nsub, int lane) {
  const int ncol = lane & 15;
  const int kb   = (lane >> 4) << 3;        // 0 or 8
  const bf16_t* p = tileT + (nsub * 16 + ncol) * 32 + kb;
  v8bf lo = *(const v8bf*)(p);
  v8bf hi = *(const v8bf*)(p + 16);
  return __builtin_shufflevector(lo, hi, 0,1,2,3,4,5,6,7,8,9,10,11,12,13,14,15);
}

__device__ __forceinline__ v8f wmma_bf16(v16bf a, v16bf b, v8f c) {
  return __builtin_amdgcn_wmma_f32_16x16x32_bf16(
      /*neg_a=*/false, a, /*neg_b=*/false, b,
      /*c_mod=*/(short)0, c, /*reuse_a=*/false, /*reuse_b=*/false);
}

// ---------------------------------------------------------------------------
// Kernel 0: weights fp32 -> bf16 (w_qkv padded from 769 to 784 rows with 0)
// ---------------------------------------------------------------------------
__global__ __launch_bounds__(256) void prep_weights_kernel(
    const float* __restrict__ wq_in, const float* __restrict__ wo_in,
    bf16_t* __restrict__ wq, bf16_t* __restrict__ wo) {
  const int i = blockIdx.x * blockDim.x + threadIdx.x;
  const int nq = OPAD * EMBED;
  if (i < nq) {
    const int r = i / EMBED;
    wq[i] = (r < OQKV) ? (bf16_t)wq_in[i] : (bf16_t)0.0f;
  } else if (i < nq + EMBED * EMBED) {
    const int j = i - nq;
    wo[j] = (bf16_t)wo_in[j];
  }
}

// ---------------------------------------------------------------------------
// Kernel 1: qkv = W_qkv @ x + b_qkv   (per slab s=(b,p): 784 x 384 x 4096)
// WG: 7 waves, tile M=112 x N=128, K-step 32, double-buffered LDS.
// ---------------------------------------------------------------------------
__global__ __launch_bounds__(224) void qkv_gemm_kernel(
    const float* __restrict__ x, const float* __restrict__ b_qkv,
    const bf16_t* __restrict__ wq, bf16_t* __restrict__ qkvp) {
  __shared__ bf16_t Wt[2][112 * 32];   // A tiles, row-major [m][k]
  __shared__ bf16_t Xt[2][128 * 32];   // B tiles, transposed [n][k]

  const int t    = threadIdx.x;
  const int lane = t & 31;
  const int wave = t >> 5;                  // 0..6 -> 16 M-rows each
  const int n0   = blockIdx.x * 128;
  const int m0   = blockIdx.y * 112;
  const int s    = blockIdx.z;
  const int b    = s >> 2, p = s & 3;

  // x[b][c][p][n] : base for c = 0 of this (b,p) slab; c stride = NP*NN
  const float* xs = x + (((size_t)b * EMBED) * NP + p) * NN;

  auto load_tiles = [&](int k0, int buf) {
    // W tile: 112 rows x 64B = 448 x 16B async transfers (2 per thread)
    for (int j = t; j < 448; j += 224) {
      const int r = j >> 2, o = (j & 3) << 3;        // o in bf16 elements
      async_copy16(wq + (size_t)(m0 + r) * EMBED + k0 + o, &Wt[buf][r * 32 + o]);
    }
    // X tile: 32(K) x 128(N) fp32 -> bf16, transposed [n][k]; two K-rows per
    // chunk so the bf16 pair along K packs into one ds_store_b32.
    for (int j = t; j < 512; j += 224) {             // chunk = 2 rows x float4
      const int R  = j >> 5;                         // K-row pair 0..15
      const int c4 = j & 31;                         // float4 within row
      const float* p0 = xs + (size_t)(k0 + 2 * R) * (NP * NN) + n0 + c4 * 4;
      const v4f a0 = *(const v4f*)p0;
      const v4f a1 = *(const v4f*)(p0 + NP * NN);
      const int nb = c4 * 4;
      bf16_t* xt = &Xt[buf][0];
      v2bf pk;
      pk.x = (bf16_t)a0.x; pk.y = (bf16_t)a1.x; *(v2bf*)&xt[(nb + 0) * 32 + 2 * R] = pk;
      pk.x = (bf16_t)a0.y; pk.y = (bf16_t)a1.y; *(v2bf*)&xt[(nb + 1) * 32 + 2 * R] = pk;
      pk.x = (bf16_t)a0.z; pk.y = (bf16_t)a1.z; *(v2bf*)&xt[(nb + 2) * 32 + 2 * R] = pk;
      pk.x = (bf16_t)a0.w; pk.y = (bf16_t)a1.w; *(v2bf*)&xt[(nb + 3) * 32 + 2 * R] = pk;
    }
  };

  v8f acc[8] = {};
  load_tiles(0, 0);
  for (int kk = 0; kk < KSTEPS; kk++) {
    async_wait_all();          // our buf[kk&1] copies are the only outstanding
    __syncthreads();           // all waves' tile writes visible
    if (kk + 1 < KSTEPS) load_tiles((kk + 1) * 32, (kk + 1) & 1);

    const int buf = kk & 1;
    const v16bf a = load_a_frag(&Wt[buf][wave * 16 * 32], lane);
    v16bf bb[8];
#pragma unroll
    for (int ns = 0; ns < 8; ns++) bb[ns] = load_b_frag(&Xt[buf][0], ns, lane);
#pragma unroll
    for (int ns = 0; ns < 8; ns++) acc[ns] = wmma_bf16(a, bb[ns], acc[ns]);
    // next-iteration barrier protects buf before it is overwritten
  }

  // Epilogue: C/D layout -> VGPR i holds (M = i + 8*(lane>=16), N = lane&15)
  const int rsub = (lane >> 4) << 3;
  const int col  = lane & 15;
#pragma unroll
  for (int i = 0; i < 8; i++) {
    const int row  = m0 + wave * 16 + rsub + i;
    const float bias = (row < OQKV) ? b_qkv[row] : 0.0f;
    bf16_t* dst = qkvp + ((size_t)s * OPAD + row) * NN + n0 + col;
#pragma unroll
    for (int ns = 0; ns < 8; ns++)
      dst[ns * 16] = (bf16_t)(acc[ns][i] + bias);
  }
}

// ---------------------------------------------------------------------------
// Kernel 2: scores = softmax(q over N); ctx[c] = sum_n k[c][n]*scores[n]
// One block per slab; scores staged in LDS (16KB).
// ---------------------------------------------------------------------------
__global__ __launch_bounds__(256) void attn_ctx_kernel(
    const bf16_t* __restrict__ qkvp, float* __restrict__ ctx) {
  __shared__ float sc[NN];
  __shared__ float red[256];
  const int t = threadIdx.x, s = blockIdx.x;
  const bf16_t* q = qkvp + (size_t)s * OPAD * NN;       // row 0 = q logits

  float vals[16];
  float m = -3.0e38f;
#pragma unroll
  for (int i = 0; i < 16; i++) {
    const float v = (float)q[t + i * 256];
    vals[i] = v;
    m = fmaxf(m, v);
  }
  red[t] = m; __syncthreads();
  for (int off = 128; off > 0; off >>= 1) {
    if (t < off) red[t] = fmaxf(red[t], red[t + off]);
    __syncthreads();
  }
  m = red[0]; __syncthreads();

  float sum = 0.0f;
#pragma unroll
  for (int i = 0; i < 16; i++) {
    const float e = __expf(vals[i] - m);
    sc[t + i * 256] = e;
    sum += e;
  }
  red[t] = sum; __syncthreads();
  for (int off = 128; off > 0; off >>= 1) {
    if (t < off) red[t] += red[t + off];
    __syncthreads();
  }
  const float inv = 1.0f / red[0]; __syncthreads();
#pragma unroll
  for (int i = 0; i < 16; i++) sc[t + i * 256] *= inv;
  __syncthreads();

  // context vector: one channel per wave at a time, wave32 reduction
  const int wave = t >> 5, lane = t & 31;
  for (int c = wave; c < EMBED; c += 8) {
    const bf16_t* krow = qkvp + ((size_t)s * OPAD + 1 + c) * NN;
    float acc = 0.0f;
    for (int i = lane; i < NN / 4; i += 32) {
      const v4f sv = *(const v4f*)&sc[i * 4];
      const bf16_t* kp = krow + i * 4;
      acc += (float)kp[0] * sv.x + (float)kp[1] * sv.y +
             (float)kp[2] * sv.z + (float)kp[3] * sv.w;
    }
#pragma unroll
    for (int off = 16; off > 0; off >>= 1) acc += __shfl_xor(acc, off, 32);
    if (lane == 0) ctx[s * EMBED + c] = acc;
  }
}

// ---------------------------------------------------------------------------
// Kernel 3: out = W_out @ (relu(v) * ctx) + b_out     (per slab 384x384x4096)
// WG: 8 waves, tile M=128 x N=128, double-buffered LDS.
// v rows live at qkvp[s][385 + c][*] (bf16).
// ---------------------------------------------------------------------------
__global__ __launch_bounds__(256) void out_gemm_kernel(
    const bf16_t* __restrict__ qkvp, const float* __restrict__ ctx,
    const bf16_t* __restrict__ wo, const float* __restrict__ b_out,
    float* __restrict__ out) {
  __shared__ bf16_t Wt[2][128 * 32];
  __shared__ bf16_t Xt[2][128 * 32];
  __shared__ float  ctxs[EMBED];

  const int t    = threadIdx.x;
  const int lane = t & 31;
  const int wave = t >> 5;                  // 0..7
  const int n0   = blockIdx.x * 128;
  const int m0   = blockIdx.y * 128;
  const int s    = blockIdx.z;
  const int b    = s >> 2, p = s & 3;

  for (int j = t; j < EMBED; j += 256) ctxs[j] = ctx[s * EMBED + j];
  __syncthreads();

  const bf16_t* vbase = qkvp + ((size_t)s * OPAD + 1 + EMBED) * NN;  // row 385

  auto load_tiles = [&](int k0, int buf) {
    // W tile: 128 rows x 64B = 512 x 16B async transfers (2 per thread)
    for (int j = t; j < 512; j += 256) {
      const int r = j >> 2, o = (j & 3) << 3;
      async_copy16(wo + (size_t)(m0 + r) * EMBED + k0 + o, &Wt[buf][r * 32 + o]);
    }
    // B operand: relu(v)*ctx, transposed [n][k]; two K-rows per chunk -> v2bf
    for (int j = t; j < 256; j += 256) {            // chunk = 2 rows x 8 n
      const int R = j >> 4, ch = j & 15;
      const bf16_t* vp = vbase + (size_t)(k0 + 2 * R) * NN + n0 + ch * 8;
      const v8bf r0 = *(const v8bf*)vp;
      const v8bf r1 = *(const v8bf*)(vp + NN);
      const float s0 = ctxs[k0 + 2 * R];
      const float s1 = ctxs[k0 + 2 * R + 1];
      bf16_t* xt = &Xt[buf][0];
#pragma unroll
      for (int e = 0; e < 8; e++) {
        v2bf pk;
        pk.x = (bf16_t)(fmaxf((float)r0[e], 0.0f) * s0);
        pk.y = (bf16_t)(fmaxf((float)r1[e], 0.0f) * s1);
        *(v2bf*)&xt[(ch * 8 + e) * 32 + 2 * R] = pk;
      }
    }
  };

  v8f acc[8] = {};
  load_tiles(0, 0);
  for (int kk = 0; kk < KSTEPS; kk++) {
    async_wait_all();
    __syncthreads();
    if (kk + 1 < KSTEPS) load_tiles((kk + 1) * 32, (kk + 1) & 1);

    const int buf = kk & 1;
    const v16bf a = load_a_frag(&Wt[buf][wave * 16 * 32], lane);
    v16bf bb[8];
#pragma unroll
    for (int ns = 0; ns < 8; ns++) bb[ns] = load_b_frag(&Xt[buf][0], ns, lane);
#pragma unroll
    for (int ns = 0; ns < 8; ns++) acc[ns] = wmma_bf16(a, bb[ns], acc[ns]);
  }

  const int rsub = (lane >> 4) << 3;
  const int col  = lane & 15;
#pragma unroll
  for (int i = 0; i < 8; i++) {
    const int row   = m0 + wave * 16 + rsub + i;
    const float bias = b_out[row];
    float* dst = out + (((size_t)(b * EMBED + row)) * NP + p) * NN + n0 + col;
#pragma unroll
    for (int ns = 0; ns < 8; ns++)
      dst[ns * 16] = acc[ns][i] + bias;
  }
}

// ---------------------------------------------------------------------------
extern "C" void kernel_launch(void* const* d_in, const int* in_sizes, int n_in,
                              void* d_out, int out_size, void* d_ws, size_t ws_size,
                              hipStream_t stream) {
  const float* x     = (const float*)d_in[0];
  const float* w_qkv = (const float*)d_in[1];
  const float* b_qkv = (const float*)d_in[2];
  const float* w_out = (const float*)d_in[3];
  const float* b_out = (const float*)d_in[4];
  float* out = (float*)d_out;

  // workspace layout (16B-aligned sections)
  char* ws = (char*)d_ws;
  const size_t wq_bytes  = (size_t)OPAD * EMBED * 2;      // 602112
  const size_t wo_bytes  = (size_t)EMBED * EMBED * 2;     // 294912
  const size_t ctx_bytes = (size_t)NSLAB * EMBED * 4;     // 49152
  bf16_t* wq   = (bf16_t*)ws;
  bf16_t* wo   = (bf16_t*)(ws + wq_bytes);
  float*  ctx  = (float*)(ws + wq_bytes + wo_bytes);
  bf16_t* qkvp = (bf16_t*)(ws + wq_bytes + wo_bytes + ctx_bytes);
  // qkvp: NSLAB * OPAD * NN * 2 bytes  (~196 MB)

  const int prep_n = OPAD * EMBED + EMBED * EMBED;
  prep_weights_kernel<<<(prep_n + 255) / 256, 256, 0, stream>>>(w_qkv, w_out, wq, wo);

  qkv_gemm_kernel<<<dim3(NN / 128, OPAD / 112, NSLAB), 224, 0, stream>>>(
      x, b_qkv, wq, qkvp);

  attn_ctx_kernel<<<NSLAB, 256, 0, stream>>>(qkvp, ctx);

  out_gemm_kernel<<<dim3(NN / 128, EMBED / 128, NSLAB), 256, 0, stream>>>(
      qkvp, ctx, wo, b_out, out);
}
